// TransNeXt_71932112273764
// MI455X (gfx1250) — compile-verified, matrix operand order checked by reading, not run on hardware
//
#include <hip/hip_runtime.h>

typedef __attribute__((ext_vector_type(16))) __bf16 v16bf;
typedef __attribute__((ext_vector_type(8)))  float  v8f;

#define NHEAD 8
#define HDIM  64
#define SEQ   1024
#define CDIM  512
#define SEQSCALE 6.93147180559945f   // ln(32*32)

// ---------------------------------------------------------------------------
// CDNA5 WMMA fragment helpers (wave32).
// A-frag (16x32 bf16, MxK): lane = {half,m}: m = lane&15 is the row,
//   element i maps to k = half*8 + i        (i<8)
//                    k = 16 + half*8 + i-8  (i>=8)   -> two contiguous 8-runs.
// B-frag (32x16 bf16, KxN): lane&15 = column n, same k mapping per element.
// C/D frag (16x16 f32): col = lane&15, row = (lane>>4)*8 + vgpr_index.
// ---------------------------------------------------------------------------
static __device__ __forceinline__ int frag_k(int lane, int i) {
  int base = ((lane >> 4) & 1) * 8;
  return (i < 8) ? (base + i) : (16 + base + (i - 8));
}

static __device__ __forceinline__ v16bf load_frag_bf16(const __bf16* __restrict__ p, int lane) {
  int base = ((lane >> 4) & 1) * 8;
  v16bf r;
#pragma unroll
  for (int i = 0; i < 8; ++i) r[i] = p[base + i];          // contiguous 16B
#pragma unroll
  for (int i = 0; i < 8; ++i) r[8 + i] = p[16 + base + i]; // contiguous 16B
  return r;
}

static __device__ __forceinline__ v16bf load_frag_f32(const float* __restrict__ p, int lane) {
  int base = ((lane >> 4) & 1) * 8;
  v16bf r;
#pragma unroll
  for (int i = 0; i < 8; ++i) r[i] = (__bf16)p[base + i];
#pragma unroll
  for (int i = 0; i < 8; ++i) r[8 + i] = (__bf16)p[16 + base + i];
  return r;
}

#define WMMA_BF16(A, B, Cacc) \
  __builtin_amdgcn_wmma_f32_16x16x32_bf16(false, (A), false, (B), (short)0, (Cacc), false, false)

// ===========================================================================
// Kernel 1: continuous-position-bias MLP  ->  bias_table[T][NHEAD]
// ===========================================================================
__global__ void cpb_mlp_kernel(const float* __restrict__ coords,  // [T,2]
                               const float* __restrict__ fc1_w,   // [512,2]
                               const float* __restrict__ fc1_b,   // [512]
                               const float* __restrict__ fc2_w,   // [NHEAD,512]
                               const float* __restrict__ fc2_b,   // [NHEAD]
                               float* __restrict__ bias_table,    // [T,NHEAD]
                               int T) {
  int g = blockIdx.x * blockDim.x + threadIdx.x;
  if (g >= T * NHEAD) return;
  int t = g / NHEAD, h = g % NHEAD;
  float c0 = coords[t * 2 + 0], c1 = coords[t * 2 + 1];
  float acc = fc2_b[h];
  for (int j = 0; j < 512; ++j) {
    float hid = fmaf(c1, fc1_w[j * 2 + 1], fmaf(c0, fc1_w[j * 2 + 0], fc1_b[j]));
    hid = fmaxf(hid, 0.0f);
    acc = fmaf(hid, fc2_w[h * 512 + j], acc);
  }
  bias_table[t * NHEAD + h] = acc;
}

// ===========================================================================
// Kernel 2: QKV GEMM (WMMA bf16) fused with l2-norm / query-embedding /
// temperature-scale epilogue; writes packed bf16 Q,K,V in [B,NH,N,HD].
// 64-wide column tile == exactly one (part, head) so each wave's 16x64 strip
// is a complete head row -> row norms reduce within the wave.
// Block = 256 thr (8 waves), tile = 128 rows x 64 cols, K-loop step 32.
// ===========================================================================
__global__ __launch_bounds__(256)
void qkv_norm_pack_kernel(const float* __restrict__ x,      // [B*SEQ, CDIM]
                          const float* __restrict__ w,      // [3*CDIM, CDIM]
                          const float* __restrict__ bvec,   // [3*CDIM]
                          const float* __restrict__ temperature, // [NHEAD]
                          const float* __restrict__ qemb,   // [NHEAD*HDIM]
                          __bf16* __restrict__ qh,
                          __bf16* __restrict__ kh,
                          __bf16* __restrict__ vh) {
  __shared__ __bf16 Bs[64][40];   // weight tile [n][k], stride 40 -> no bank conflicts
  const int tid = threadIdx.x, wave = tid >> 5, lane = tid & 31;
  const int j0 = blockIdx.x * 64;          // output col base (part/head aligned)
  const int row0 = blockIdx.y * 128;       // output row base (b*n flat)
  const int m = lane & 15;
  const float* arow = &x[(size_t)(row0 + wave * 16 + m) * CDIM];

  v8f acc[4] = {};
  for (int k0 = 0; k0 < CDIM; k0 += 32) {
    __syncthreads();                       // WAR: prior frag reads done
#pragma unroll
    for (int t = 0; t < 2; ++t) {          // stage 64x32 fp32 weights -> bf16 LDS
      int f = t * 256 + tid;
      int n = f >> 3, c4 = (f & 7) * 4;
      const float4 v4 = *(const float4*)&w[(size_t)(j0 + n) * CDIM + k0 + c4];
      Bs[n][c4 + 0] = (__bf16)v4.x; Bs[n][c4 + 1] = (__bf16)v4.y;
      Bs[n][c4 + 2] = (__bf16)v4.z; Bs[n][c4 + 3] = (__bf16)v4.w;
    }
    __syncthreads();
    v16bf a = load_frag_f32(arow + k0, lane);
#pragma unroll
    for (int f = 0; f < 4; ++f) {
      v16bf bb = load_frag_bf16(&Bs[f * 16 + m][0], lane);
      acc[f] = WMMA_BF16(a, bb, acc[f]);
    }
  }

  // ---- epilogue -----------------------------------------------------------
  const int col = m;
  const int part = j0 / CDIM;              // 0=q 1=k 2=v
  const int h = (j0 % CDIM) / HDIM;
  const int rbase = (lane >> 4) * 8;
#pragma unroll
  for (int f = 0; f < 4; ++f) {
    float bv = bvec[j0 + f * 16 + col];
#pragma unroll
    for (int j = 0; j < 8; ++j) acc[f][j] += bv;
  }
  float rs[8];
#pragma unroll
  for (int j = 0; j < 8; ++j) rs[j] = 1.0f;
  if (part != 2) {                         // row l2 norm over the full head (64 wide)
#pragma unroll
    for (int j = 0; j < 8; ++j) {
      float ss = 0.0f;
#pragma unroll
      for (int f = 0; f < 4; ++f) ss = fmaf(acc[f][j], acc[f][j], ss);
#pragma unroll
      for (int d = 1; d < 16; d <<= 1) ss += __shfl_xor(ss, d, 32);
      rs[j] = 1.0f / fmaxf(sqrtf(ss), 1e-12f);
    }
  }
  const float hscale = log1pf(__expf(temperature[h])) * SEQSCALE;  // softplus * ln(1024)
#pragma unroll
  for (int f = 0; f < 4; ++f) {
    int hd = f * 16 + col;
#pragma unroll
    for (int j = 0; j < 8; ++j) {
      int row = row0 + wave * 16 + rbase + j;
      int b = row / SEQ, n = row % SEQ;
      size_t dst = (((size_t)(b * NHEAD + h)) * SEQ + n) * HDIM + hd;
      float val = acc[f][j];
      if (part == 0) {
        qh[dst] = (__bf16)((val * rs[j] + qemb[h * HDIM + hd]) * hscale);
      } else if (part == 1) {
        kh[dst] = (__bf16)(val * rs[j]);
      } else {
        vh[dst] = (__bf16)val;
      }
    }
  }
}

// ===========================================================================
// Kernel 3: flash attention per (b,h). Block = 128 thr (4 waves); each wave
// owns 16 query rows (64/block). Online softmax over 32-key blocks:
//   S(16x32) = 4 wmma, bias gather, row-stats via shfl, P->LDS->A-frag,
//   O += P·V = 4 wmma.  Never materializes the [N,N] attention matrix.
// ===========================================================================
__global__ __launch_bounds__(128)
void flash_attn_kernel(const __bf16* __restrict__ qh,
                       const __bf16* __restrict__ kh,
                       const __bf16* __restrict__ vh,
                       const float* __restrict__ bias_table, // [T,NHEAD]
                       const int* __restrict__ rel_idx,      // [SEQ*SEQ]
                       __bf16* __restrict__ attn_out) {      // [B*SEQ, CDIM] bf16
  __shared__ __bf16 Plds[4][16][40];       // per-wave P transpose staging
  const int bh = blockIdx.x;
  const int b = bh / NHEAD, h = bh % NHEAD;
  const int tid = threadIdx.x, wave = tid >> 5, lane = tid & 31;
  const int m = lane & 15, col = lane & 15, rbase = (lane >> 4) * 8;
  const int q0 = blockIdx.y * 64 + wave * 16;

  const __bf16* Qb = qh + (size_t)bh * SEQ * HDIM;
  const __bf16* Kb = kh + (size_t)bh * SEQ * HDIM;
  const __bf16* Vb = vh + (size_t)bh * SEQ * HDIM;

  // Q fragments for the two 32-wide hd chunks (reused over all key blocks)
  v16bf qf0 = load_frag_bf16(&Qb[(size_t)(q0 + m) * HDIM + 0], lane);
  v16bf qf1 = load_frag_bf16(&Qb[(size_t)(q0 + m) * HDIM + 32], lane);

  float mrow[8], lrow[8];
  v8f o[4] = {};
#pragma unroll
  for (int j = 0; j < 8; ++j) { mrow[j] = -1e30f; lrow[j] = 0.0f; }

  for (int kb = 0; kb < SEQ / 32; ++kb) {
    const int key0 = kb * 32;
    v8f s0 = {}, s1 = {};
    {   // S = Q·K^T over both 16-key tiles, K=32 hd chunks
      v16bf kf = load_frag_bf16(&Kb[(size_t)(key0 + m) * HDIM + 0], lane);
      s0 = WMMA_BF16(qf0, kf, s0);
      kf = load_frag_bf16(&Kb[(size_t)(key0 + m) * HDIM + 32], lane);
      s0 = WMMA_BF16(qf1, kf, s0);
      kf = load_frag_bf16(&Kb[(size_t)(key0 + 16 + m) * HDIM + 0], lane);
      s1 = WMMA_BF16(qf0, kf, s1);
      kf = load_frag_bf16(&Kb[(size_t)(key0 + 16 + m) * HDIM + 32], lane);
      s1 = WMMA_BF16(qf1, kf, s1);
    }
    // relative position bias gather (index array is 4MB -> L2 resident)
#pragma unroll
    for (int j = 0; j < 8; ++j) {
      int qrow = q0 + rbase + j;
      int i0 = rel_idx[qrow * SEQ + key0 + col];
      int i1 = rel_idx[qrow * SEQ + key0 + 16 + col];
      s0[j] += bias_table[i0 * NHEAD + h];
      s1[j] += bias_table[i1 * NHEAD + h];
    }
    // online softmax update
    float corr[8];
#pragma unroll
    for (int j = 0; j < 8; ++j) {
      float mx = fmaxf(s0[j], s1[j]);
#pragma unroll
      for (int d = 1; d < 16; d <<= 1) mx = fmaxf(mx, __shfl_xor(mx, d, 32));
      float mn = fmaxf(mrow[j], mx);
      corr[j] = __expf(mrow[j] - mn);
      mrow[j] = mn;
      float p0 = __expf(s0[j] - mn);
      float p1 = __expf(s1[j] - mn);
      s0[j] = p0; s1[j] = p1;
      float rsum = p0 + p1;
#pragma unroll
      for (int d = 1; d < 16; d <<= 1) rsum += __shfl_xor(rsum, d, 32);
      lrow[j] = lrow[j] * corr[j] + rsum;
    }
#pragma unroll
    for (int f = 0; f < 4; ++f)
#pragma unroll
      for (int j = 0; j < 8; ++j) o[f][j] *= corr[j];

    // transpose P (C-frag layout) -> A-frag layout through per-wave LDS
#pragma unroll
    for (int j = 0; j < 8; ++j) {
      Plds[wave][rbase + j][col]      = (__bf16)s0[j];
      Plds[wave][rbase + j][16 + col] = (__bf16)s1[j];
    }
    __syncthreads();   // order LDS write->read (also keeps waves in lockstep)
    v16bf pf = load_frag_bf16(&Plds[wave][m][0], lane);

    // O += P · V : V as B-frag (k = key, n = hd)
#pragma unroll
    for (int f = 0; f < 4; ++f) {
      v16bf vf;
#pragma unroll
      for (int i = 0; i < 16; ++i)
        vf[i] = Vb[(size_t)(key0 + frag_k(lane, i)) * HDIM + f * 16 + col];
      o[f] = WMMA_BF16(pf, vf, o[f]);
    }
    __syncthreads();   // protect Plds against next iteration's writes
  }

  // finalize: O / l, store bf16 in [B,N, h*HD+hd] for the projection GEMM
#pragma unroll
  for (int f = 0; f < 4; ++f) {
    int hd = f * 16 + col;
#pragma unroll
    for (int j = 0; j < 8; ++j) {
      int row = q0 + rbase + j;
      float val = o[f][j] / lrow[j];
      attn_out[((size_t)b * SEQ + row) * CDIM + h * HDIM + hd] = (__bf16)val;
    }
  }
}

// ===========================================================================
// Kernel 4: output projection GEMM (WMMA bf16 -> fp32 + bias)
// ===========================================================================
__global__ __launch_bounds__(256)
void proj_gemm_kernel(const __bf16* __restrict__ a,   // [B*SEQ, CDIM] bf16
                      const float* __restrict__ w,    // [CDIM, CDIM]
                      const float* __restrict__ bvec, // [CDIM]
                      float* __restrict__ out) {      // [B*SEQ, CDIM] fp32
  __shared__ __bf16 Bs[64][40];
  const int tid = threadIdx.x, wave = tid >> 5, lane = tid & 31;
  const int j0 = blockIdx.x * 64;
  const int row0 = blockIdx.y * 128;
  const int m = lane & 15, col = lane & 15, rbase = (lane >> 4) * 8;
  const __bf16* arow = &a[(size_t)(row0 + wave * 16 + m) * CDIM];

  v8f acc[4] = {};
  for (int k0 = 0; k0 < CDIM; k0 += 32) {
    __syncthreads();
#pragma unroll
    for (int t = 0; t < 2; ++t) {
      int f = t * 256 + tid;
      int n = f >> 3, c4 = (f & 7) * 4;
      const float4 v4 = *(const float4*)&w[(size_t)(j0 + n) * CDIM + k0 + c4];
      Bs[n][c4 + 0] = (__bf16)v4.x; Bs[n][c4 + 1] = (__bf16)v4.y;
      Bs[n][c4 + 2] = (__bf16)v4.z; Bs[n][c4 + 3] = (__bf16)v4.w;
    }
    __syncthreads();
    v16bf af = load_frag_bf16(arow + k0, lane);
#pragma unroll
    for (int f = 0; f < 4; ++f) {
      v16bf bb = load_frag_bf16(&Bs[f * 16 + m][0], lane);
      acc[f] = WMMA_BF16(af, bb, acc[f]);
    }
  }
#pragma unroll
  for (int f = 0; f < 4; ++f) {
    float bv = bvec[j0 + f * 16 + col];
#pragma unroll
    for (int j = 0; j < 8; ++j) {
      int row = row0 + wave * 16 + rbase + j;
      out[(size_t)row * CDIM + j0 + f * 16 + col] = acc[f][j] + bv;
    }
  }
}

// ===========================================================================
extern "C" void kernel_launch(void* const* d_in, const int* in_sizes, int n_in,
                              void* d_out, int out_size, void* d_ws, size_t ws_size,
                              hipStream_t stream) {
  const float* x      = (const float*)d_in[0];
  const float* qkv_w  = (const float*)d_in[1];
  const float* qkv_b  = (const float*)d_in[2];
  const float* temp   = (const float*)d_in[3];
  const float* qemb   = (const float*)d_in[4];
  const float* proj_w = (const float*)d_in[5];
  const float* proj_b = (const float*)d_in[6];
  const float* fc1w   = (const float*)d_in[7];
  const float* fc1b   = (const float*)d_in[8];
  const float* fc2w   = (const float*)d_in[9];
  const float* fc2b   = (const float*)d_in[10];
  const float* coords = (const float*)d_in[11];
  const int*   relidx = (const int*)d_in[12];

  const int T  = in_sizes[11] / 2;        // unique relative coords (3969)
  const int BN = in_sizes[0] / CDIM;      // B*N = 32768
  const int B  = BN / SEQ;                // 32

  // workspace layout (all bf16 intermediates -> ~128MB, L2-resident mid-pipe)
  __bf16* qh = (__bf16*)d_ws;
  const size_t hsz = (size_t)B * NHEAD * SEQ * HDIM;
  __bf16* kh   = qh + hsz;
  __bf16* vh   = kh + hsz;
  __bf16* attn = vh + hsz;
  float* bias_table = (float*)(attn + (size_t)BN * CDIM);

  cpb_mlp_kernel<<<(T * NHEAD + 255) / 256, 256, 0, stream>>>(
      coords, fc1w, fc1b, fc2w, fc2b, bias_table, T);

  qkv_norm_pack_kernel<<<dim3(3 * CDIM / 64, BN / 128), 256, 0, stream>>>(
      x, qkv_w, qkv_b, temp, qemb, qh, kh, vh);

  flash_attn_kernel<<<dim3(B * NHEAD, SEQ / 64), 128, 0, stream>>>(
      qh, kh, vh, bias_table, relidx, attn);

  proj_gemm_kernel<<<dim3(CDIM / 64, BN / 128), 256, 0, stream>>>(
      attn, proj_w, proj_b, (float*)d_out);
}
